// W8A8BFP32OFP32Linear_541165879315
// MI455X (gfx1250) — 8-bit matrix operand order repaired and checked by reading, compiled, not run on hardware
//
#include <hip/hip_runtime.h>
#include <hip/hip_bf16.h>
#include <stdint.h>


typedef __attribute__((ext_vector_type(8))) int v8i;
typedef __attribute__((ext_vector_type(4))) int v4i;
typedef __attribute__((ext_vector_type(2))) int v2i;

typedef __attribute__((address_space(1))) v4i* gv4i_p;
typedef __attribute__((address_space(3))) v4i* lv4i_p;

#define M_TOT 8192
#define N_TOT 4096
#define K_TOT 4096
#define BM 128
#define BN 128
#define BK 64
#define NKT (K_TOT / BK)
#define LDSP 80
#define TILE_BYTES (BM * LDSP)
#define NBUF 3

#if __has_builtin(__builtin_amdgcn_global_load_async_to_lds_b128) && \
    __has_builtin(__builtin_amdgcn_s_wait_asynccnt)
#define USE_ASYNC 1
#endif

__global__ __launch_bounds__(256)
void w8a8_bias_gemm(const char* __restrict__ X, const char* __restrict__ W,
                    const float* __restrict__ bias, const float* __restrict__ alpha_p,
                    float* __restrict__ Y)
{
    __shared__ __attribute__((aligned(16))) char smem[NBUF * 2 * TILE_BYTES];

    const int tid   = threadIdx.x;
    const int lane  = tid & 31;
    const int wid   = tid >> 5;
    const int wm    = wid & 3;
    const int wn    = wid >> 2;
    const int nl    = lane & 15;
    const int khalf = lane >> 4;

    const int gm0 = blockIdx.y * BM;
    const int gn0 = blockIdx.x * BN;

    const int crow = tid >> 2;
    const int ccol = (tid & 3) * 16;

    const char* gA = X + (size_t)(gm0 + crow) * K_TOT + ccol;
    const char* gB = W + (size_t)(gn0 + crow) * K_TOT + ccol;
    const size_t rowStep = (size_t)64 * K_TOT;

    const int lOff0 = crow * LDSP + ccol;
    const int lOff1 = (crow + 64) * LDSP + ccol;

    v8i acc[2][4];
#pragma unroll
    for (int mi = 0; mi < 2; ++mi)
#pragma unroll
        for (int ni = 0; ni < 4; ++ni)
#pragma unroll
            for (int e = 0; e < 8; ++e) acc[mi][ni][e] = 0;

#ifdef USE_ASYNC
    auto prefetch = [&](int kt, int buf) {
        const char* pa = gA + (size_t)kt * BK;
        const char* pb = gB + (size_t)kt * BK;
        char* l = smem + buf * (2 * TILE_BYTES);
        __builtin_amdgcn_global_load_async_to_lds_b128(
            (gv4i_p)pa,             (lv4i_p)(l + lOff0), 0, 0);
        __builtin_amdgcn_global_load_async_to_lds_b128(
            (gv4i_p)(pa + rowStep), (lv4i_p)(l + lOff1), 0, 0);
        __builtin_amdgcn_global_load_async_to_lds_b128(
            (gv4i_p)pb,             (lv4i_p)(l + TILE_BYTES + lOff0), 0, 0);
        __builtin_amdgcn_global_load_async_to_lds_b128(
            (gv4i_p)(pb + rowStep), (lv4i_p)(l + TILE_BYTES + lOff1), 0, 0);
    };
    prefetch(0, 0);
    prefetch(1, 1);
    int curBuf = 0;
#else
    v4i ra0 = *(const v4i*)gA;
    v4i ra1 = *(const v4i*)(gA + rowStep);
    v4i rb0 = *(const v4i*)gB;
    v4i rb1 = *(const v4i*)(gB + rowStep);
#endif

    for (int kt = 0; kt < NKT; ++kt) {
#ifdef USE_ASYNC
        if (kt + 1 < NKT) __builtin_amdgcn_s_wait_asynccnt(4);
        else              __builtin_amdgcn_s_wait_asynccnt(0);
        __syncthreads();
        if (kt + 2 < NKT) prefetch(kt + 2, curBuf == 0 ? 2 : curBuf - 1);
        const char* lA = smem + curBuf * (2 * TILE_BYTES);
#else
        const int cur = kt & 1;
        {
            char* l = smem + cur * (2 * TILE_BYTES);
            *(v4i*)(l + lOff0)              = ra0;
            *(v4i*)(l + lOff1)              = ra1;
            *(v4i*)(l + TILE_BYTES + lOff0) = rb0;
            *(v4i*)(l + TILE_BYTES + lOff1) = rb1;
        }
        __syncthreads();
        if (kt + 1 < NKT) {
            const char* pa = gA + (size_t)(kt + 1) * BK;
            const char* pb = gB + (size_t)(kt + 1) * BK;
            ra0 = *(const v4i*)pa;  ra1 = *(const v4i*)(pa + rowStep);
            rb0 = *(const v4i*)pb;  rb1 = *(const v4i*)(pb + rowStep);
        }
        const char* lA = smem + cur * (2 * TILE_BYTES);
#endif
        const char* lB = lA + TILE_BYTES;

        v8i afr[2];
#pragma unroll
        for (int mi = 0; mi < 2; ++mi) {
            const char* rp = lA + (wm * 32 + mi * 16 + nl) * LDSP;
#pragma unroll
            for (int g = 0; g < 4; ++g) {
                v2i t = *(const v2i*)(rp + (g >> 1) * 32 + khalf * 16 + (g & 1) * 8);
                afr[mi][2 * g]     = t[0];
                afr[mi][2 * g + 1] = t[1];
            }
        }
        v8i bfr[4];
#pragma unroll
        for (int ni = 0; ni < 4; ++ni) {
            const char* cp = lB + (wn * 64 + ni * 16 + nl) * LDSP + khalf * 16;
            v4i lo = *(const v4i*)cp;
            v4i hi = *(const v4i*)(cp + 32);
            bfr[ni][0] = lo[0]; bfr[ni][1] = lo[1]; bfr[ni][2] = lo[2]; bfr[ni][3] = lo[3];
            bfr[ni][4] = hi[0]; bfr[ni][5] = hi[1]; bfr[ni][6] = hi[2]; bfr[ni][7] = hi[3];
        }
#pragma unroll
        for (int mi = 0; mi < 2; ++mi)
#pragma unroll
            for (int ni = 0; ni < 4; ++ni)
                acc[mi][ni] = __builtin_amdgcn_wmma_i32_16x16x64_iu8(
                    true, afr[mi], true, bfr[ni], acc[mi][ni], false, false);

#ifdef USE_ASYNC
        curBuf = (curBuf == NBUF - 1) ? 0 : curBuf + 1;
#else
        __syncthreads();
#endif
    }

    const float alpha = alpha_p[0];
    float bv[4];
    int   ncol[4];
#pragma unroll
    for (int ni = 0; ni < 4; ++ni) {
        ncol[ni] = gn0 + wn * 64 + ni * 16 + nl;
        bv[ni]   = bias[ncol[ni]];
    }
#pragma unroll
    for (int mi = 0; mi < 2; ++mi) {
#pragma unroll
        for (int r = 0; r < 8; ++r) {
            const int mg = gm0 + wm * 32 + mi * 16 + khalf * 8 + r;
            float* yr = Y + (size_t)mg * N_TOT;
#pragma unroll
            for (int ni = 0; ni < 4; ++ni)
                yr[ncol[ni]] = alpha * (float)acc[mi][ni][r] + bv[ni];
        }
    }
}

extern "C" void kernel_launch(void* const* d_in, const int* in_sizes, int n_in,
                              void* d_out, int out_size, void* d_ws, size_t ws_size,
                              hipStream_t stream) {
    const char*  X     = (const char*)d_in[0];
    const char*  Wt    = (const char*)d_in[1];
    const float* bias  = (const float*)d_in[2];
    const float* alpha = (const float*)d_in[3];
    float*       Y     = (float*)d_out;

    dim3 grid(N_TOT / BN, M_TOT / BM);
    w8a8_bias_gemm<<<grid, 256, 0, stream>>>(X, Wt, bias, alpha, Y);
}
